// GAT_26199300505825
// MI455X (gfx1250) — compile-verified, matrix operand order checked by reading, not run on hardware
//
#include <hip/hip_runtime.h>
#include <hip/hip_bf16.h>
#include <stdint.h>

#define N_NODES 50000
#define DEG     16
#define IN_DIM  256
#define HID     64
#define CLS     40
#define SLOPE   0.2f

typedef __attribute__((ext_vector_type(16))) __bf16 v16bf;
typedef __attribute__((ext_vector_type(8)))  __bf16 v8bf;
typedef __attribute__((ext_vector_type(8)))  float  v8f;

// ---------------------------------------------------------------------------
// fp32 -> bf16 elementwise convert (activations)
// ---------------------------------------------------------------------------
__global__ void __launch_bounds__(256)
f32_to_bf16_kernel(const float* __restrict__ in, __bf16* __restrict__ out, int count) {
    int i = blockIdx.x * blockDim.x + threadIdx.x;
    if (i < count) out[i] = (__bf16)in[i];
}

// ---------------------------------------------------------------------------
// Weight convert + transpose: W[K][M] fp32  ->  Wt[M][K] bf16
// (transposed so WMMA B fragments become contiguous 16B loads along K)
// ---------------------------------------------------------------------------
__global__ void __launch_bounds__(256)
w_to_bf16_t_kernel(const float* __restrict__ W, __bf16* __restrict__ Wt, int K, int M) {
    int i = blockIdx.x * blockDim.x + threadIdx.x;
    if (i < K * M) {
        int k = i / M, m = i % M;
        Wt[(size_t)m * K + k] = (__bf16)W[i];
    }
}

// ---------------------------------------------------------------------------
// WMMA GEMM:  C[nrows x M] (f32) = A[nrows x K] (bf16, row-major)
//                                  x Bt[M x K]   (bf16, row-major = W^T)
// One wave32 per 16x16 output tile; K stepped by 32 with
// v_wmma_f32_16x16x32_bf16.
//
// A fragment layout (16-bit A 16x32, ISA 7.12.2):
//   lanes 0-15 : row = lane,    elems 0-7 = K k0+0..7,  elems 8-15 = K k0+16..23
//   lanes 16-31: row = lane-16, elems 0-7 = K k0+8..15, elems 8-15 = K k0+24..31
// B fragment mirrors A with lane -> output column (Bt row-major makes both
// fragments two global_load_b128 per lane).
// C/D layout: VGPR r, lanes 0-15 -> row r, col lane; lanes 16-31 -> row r+8.
// ---------------------------------------------------------------------------
__global__ void __launch_bounds__(256)
wmma_gemm_bf16_kernel(const __bf16* __restrict__ A,
                      const __bf16* __restrict__ Bt,
                      float* __restrict__ C,
                      int nrows, int K, int M) {
    const int lane  = threadIdx.x & 31;
    const int wave  = threadIdx.x >> 5;
    const int tilesM = M >> 4;
    const int totalTiles = (nrows >> 4) * tilesM;
    const int tile = blockIdx.x * (blockDim.x >> 5) + wave;
    if (tile >= totalTiles) return;   // wave-uniform: EXEC stays all-ones

    const int mt = tile % tilesM;
    const int nt = tile / tilesM;
    const int half = lane >> 4;       // 0: lanes 0-15, 1: lanes 16-31
    const int l15  = lane & 15;

    const __bf16* arow = A  + (size_t)(nt * 16 + l15) * K + half * 8;
    const __bf16* brow = Bt + (size_t)(mt * 16 + l15) * K + half * 8;

    v8f acc = {};
    for (int k0 = 0; k0 < K; k0 += 32) {
        v8bf a_lo = *(const v8bf*)(arow + k0);
        v8bf a_hi = *(const v8bf*)(arow + k0 + 16);
        v8bf b_lo = *(const v8bf*)(brow + k0);
        v8bf b_hi = *(const v8bf*)(brow + k0 + 16);
        v16bf a, b;
#pragma unroll
        for (int i = 0; i < 8; ++i) {
            a[i] = a_lo[i]; a[8 + i] = a_hi[i];
            b[i] = b_lo[i]; b[8 + i] = b_hi[i];
        }
        // (neg_a, A, neg_b, B, c_mod, C, reuse_a, reuse_b)
        acc = __builtin_amdgcn_wmma_f32_16x16x32_bf16(
                  false, a, false, b, (short)0, acc, false, false);
    }

    float* crow = C + (size_t)(nt * 16 + half * 8) * M + mt * 16 + l15;
#pragma unroll
    for (int r = 0; r < 8; ++r)
        crow[(size_t)r * M] = acc[r];
}

// ---------------------------------------------------------------------------
// Attention scores: att_row[n][h] = <feat[n,h,:], al[h,:]>, same for att_col.
// blockDim.x == H*D (<= 256); one block per node.
// ---------------------------------------------------------------------------
__global__ void att_scores_kernel(const float* __restrict__ feat,
                                  const float* __restrict__ al,
                                  const float* __restrict__ ar,
                                  float* __restrict__ att_row,
                                  float* __restrict__ att_col,
                                  int H, int D) {
    __shared__ float pl[256];
    __shared__ float pr[256];
    const int n = blockIdx.x;
    const int t = threadIdx.x;
    const int F = H * D;
    float f = feat[(size_t)n * F + t];
    pl[t] = f * al[t];
    pr[t] = f * ar[t];
    __syncthreads();
    if (t < H) {
        float sl = 0.f, sr = 0.f;
        for (int d = 0; d < D; ++d) { sl += pl[t * D + d]; sr += pr[t * D + d]; }
        att_row[(size_t)n * H + t] = sl;
        att_col[(size_t)n * H + t] = sr;
    }
}

// ---------------------------------------------------------------------------
// Edge softmax + aggregation, layers 0/1: F=256, H=4, D=64.
// One 256-thread block per destination node (fixed degree 16).
// ---------------------------------------------------------------------------
__global__ void __launch_bounds__(256)
agg256_kernel(const int* __restrict__ col_ind,
              const float* __restrict__ feat,
              const float* __restrict__ att_row,
              const float* __restrict__ att_col,
              float* __restrict__ out) {
    __shared__ int   s_src[DEG];
    __shared__ float s_lg[DEG * 4];   // [h][e]
    __shared__ float s_mx[4], s_inv[4];
    const int n = blockIdx.x;
    const int t = threadIdx.x;

    if (t < DEG) s_src[t] = col_ind[n * DEG + t];
    __syncthreads();
    if (t < DEG * 4) {
        const int e = t & 15, h = t >> 4;
        float lg = att_row[(size_t)n * 4 + h] + att_col[(size_t)s_src[e] * 4 + h];
        s_lg[t] = lg > 0.f ? lg : SLOPE * lg;
    }
    __syncthreads();
    if (t < 4) {
        float m = -1e30f;
        for (int e = 0; e < DEG; ++e) m = fmaxf(m, s_lg[t * DEG + e]);
        float s = 0.f;
        for (int e = 0; e < DEG; ++e) s += __expf(s_lg[t * DEG + e] - m);
        s_mx[t] = m; s_inv[t] = 1.f / s;
    }
    __syncthreads();
    if (t < DEG * 4) {
        const int h = t >> 4;
        s_lg[t] = __expf(s_lg[t] - s_mx[h]) * s_inv[h];   // alpha
    }
    __syncthreads();
    const int h = t >> 6;  // D = 64
    float acc = 0.f;
#pragma unroll
    for (int e = 0; e < DEG; ++e)
        acc += s_lg[h * DEG + e] * feat[(size_t)s_src[e] * 256 + t];
    out[(size_t)n * 256 + t] = acc;
}

// ---------------------------------------------------------------------------
// Final layer: F=240, H=6, D=CLS=40, then mean over heads -> out[n][40].
// ---------------------------------------------------------------------------
__global__ void __launch_bounds__(256)
agg_final_kernel(const int* __restrict__ col_ind,
                 const float* __restrict__ feat,
                 const float* __restrict__ att_row,
                 const float* __restrict__ att_col,
                 float* __restrict__ out) {
    __shared__ int   s_src[DEG];
    __shared__ float s_lg[DEG * 6];
    __shared__ float s_mx[6], s_inv[6];
    __shared__ float s_acc[240];
    const int n = blockIdx.x;
    const int t = threadIdx.x;

    if (t < DEG) s_src[t] = col_ind[n * DEG + t];
    __syncthreads();
    if (t < DEG * 6) {
        const int e = t & 15, h = t >> 4;
        float lg = att_row[(size_t)n * 6 + h] + att_col[(size_t)s_src[e] * 6 + h];
        s_lg[t] = lg > 0.f ? lg : SLOPE * lg;
    }
    __syncthreads();
    if (t < 6) {
        float m = -1e30f;
        for (int e = 0; e < DEG; ++e) m = fmaxf(m, s_lg[t * DEG + e]);
        float s = 0.f;
        for (int e = 0; e < DEG; ++e) s += __expf(s_lg[t * DEG + e] - m);
        s_mx[t] = m; s_inv[t] = 1.f / s;
    }
    __syncthreads();
    if (t < DEG * 6) {
        const int h = t >> 4;
        s_lg[t] = __expf(s_lg[t] - s_mx[h]) * s_inv[h];
    }
    __syncthreads();
    if (t < 240) {
        const int h = t / CLS;
        float acc = 0.f;
#pragma unroll
        for (int e = 0; e < DEG; ++e)
            acc += s_lg[h * DEG + e] * feat[(size_t)s_src[e] * 240 + t];
        s_acc[t] = acc;
    }
    __syncthreads();
    if (t < CLS) {
        float o = 0.f;
#pragma unroll
        for (int h = 0; h < 6; ++h) o += s_acc[h * CLS + t];
        out[(size_t)n * CLS + t] = o * (1.f / 6.f);
    }
}

// ---------------------------------------------------------------------------
// Host launch
// ---------------------------------------------------------------------------
extern "C" void kernel_launch(void* const* d_in, const int* in_sizes, int n_in,
                              void* d_out, int out_size, void* d_ws, size_t ws_size,
                              hipStream_t stream) {
    (void)in_sizes; (void)n_in; (void)out_size; (void)ws_size;

    const int*   col_ind = (const int*)  d_in[1];
    const float* inputs  = (const float*)d_in[4];
    const float* W0  = (const float*)d_in[5];
    const float* al0 = (const float*)d_in[6];
    const float* ar0 = (const float*)d_in[7];
    const float* W1  = (const float*)d_in[8];
    const float* al1 = (const float*)d_in[9];
    const float* ar1 = (const float*)d_in[10];
    const float* W2  = (const float*)d_in[11];
    const float* al2 = (const float*)d_in[12];
    const float* ar2 = (const float*)d_in[13];
    float* out = (float*)d_out;

    // Carve workspace (256B aligned slabs; ~131 MB total).
    uintptr_t p = (uintptr_t)d_ws;
    auto carve = [&](size_t bytes) -> void* {
        p = (p + 255) & ~(uintptr_t)255;
        void* r = (void*)p; p += bytes; return r;
    };
    __bf16* hb    = (__bf16*)carve((size_t)N_NODES * 256 * sizeof(__bf16));
    float*  feat  = (float*) carve((size_t)N_NODES * 256 * sizeof(float));
    float*  htmp  = (float*) carve((size_t)N_NODES * 256 * sizeof(float));
    __bf16* W0t   = (__bf16*)carve((size_t)256 * 256 * sizeof(__bf16));
    __bf16* W1t   = (__bf16*)carve((size_t)256 * 256 * sizeof(__bf16));
    __bf16* W2t   = (__bf16*)carve((size_t)240 * 256 * sizeof(__bf16));
    float*  attr  = (float*) carve((size_t)N_NODES * 6 * sizeof(float));
    float*  attc  = (float*) carve((size_t)N_NODES * 6 * sizeof(float));

    const int actElems = N_NODES * 256;
    const int cvtBlocks = (actElems + 255) / 256;

    // Weight conversions (once per call; cheap).
    w_to_bf16_t_kernel<<<(256 * 256 + 255) / 256, 256, 0, stream>>>(W0, W0t, 256, 256);
    w_to_bf16_t_kernel<<<(256 * 256 + 255) / 256, 256, 0, stream>>>(W1, W1t, 256, 256);
    w_to_bf16_t_kernel<<<(256 * 240 + 255) / 256, 256, 0, stream>>>(W2, W2t, 256, 240);

    const int wavesPerBlock = 8;  // 256 threads
    auto gemmGrid = [&](int nrows, int M) {
        int tiles = (nrows / 16) * (M / 16);
        return (tiles + wavesPerBlock - 1) / wavesPerBlock;
    };

    // ---- Layer 0 ----
    f32_to_bf16_kernel<<<cvtBlocks, 256, 0, stream>>>(inputs, hb, actElems);
    wmma_gemm_bf16_kernel<<<gemmGrid(N_NODES, 256), 256, 0, stream>>>(hb, W0t, feat, N_NODES, 256, 256);
    att_scores_kernel<<<N_NODES, 256, 0, stream>>>(feat, al0, ar0, attr, attc, 4, 64);
    agg256_kernel<<<N_NODES, 256, 0, stream>>>(col_ind, feat, attr, attc, htmp);

    // ---- Layer 1 ----
    f32_to_bf16_kernel<<<cvtBlocks, 256, 0, stream>>>(htmp, hb, actElems);
    wmma_gemm_bf16_kernel<<<gemmGrid(N_NODES, 256), 256, 0, stream>>>(hb, W1t, feat, N_NODES, 256, 256);
    att_scores_kernel<<<N_NODES, 256, 0, stream>>>(feat, al1, ar1, attr, attc, 4, 64);
    agg256_kernel<<<N_NODES, 256, 0, stream>>>(col_ind, feat, attr, attc, htmp);

    // ---- Layer 2 (output) ----
    f32_to_bf16_kernel<<<cvtBlocks, 256, 0, stream>>>(htmp, hb, actElems);
    wmma_gemm_bf16_kernel<<<gemmGrid(N_NODES, 240), 256, 0, stream>>>(hb, W2t, feat, N_NODES, 256, 240);
    att_scores_kernel<<<N_NODES, 240, 0, stream>>>(feat, al2, ar2, attr, attc, 6, 40);
    agg_final_kernel<<<N_NODES, 256, 0, stream>>>(col_ind, feat, attr, attc, out);
}
